// SirenGINet_80719615361068
// MI455X (gfx1250) — matrix operand order repaired and checked by reading, not compiled, not run on hardware
//
#include <hip/hip_runtime.h>
#include <hip/hip_bf16.h>

typedef _Float16 half_t;
typedef __attribute__((ext_vector_type(16))) _Float16 v16h;
typedef __attribute__((ext_vector_type(8)))  _Float16 v8h;
typedef __attribute__((ext_vector_type(8)))  float    v8f;

#define WMMA_F16(A, B, C) \
    __builtin_amdgcn_wmma_f32_16x16x32_f16(false, (A), false, (B), (short)0, (C), false, false)

// ---------------------------------------------------------------------------
// Tile loaders matching CDNA5 WMMA VGPR layouts (CDNA5 ISA §7.12.2)
// A 16-bit 16x32: lane l -> row m0+(l&15); halves 0..7 = K[k0+ksel*8 .. +7],
//                 halves 8..15 = K[k0+16+ksel*8 .. +7]   (ksel = l>>4)
// ---------------------------------------------------------------------------
__device__ __forceinline__ v16h load_tileA(const half_t* base, int stride,
                                           int m0, int k0, int lane) {
    const int  r    = m0 + (lane & 15);
    const int  ksel = lane >> 4;
    const half_t* p = base + r * stride + k0 + ksel * 8;
    v8h lo = *(const v8h*)(p);
    v8h hi = *(const v8h*)(p + 16);
    v16h a;
#pragma unroll
    for (int i = 0; i < 8; ++i) { a[i] = lo[i]; a[i + 8] = hi[i]; }
    return a;
}

// B 16-bit 32x16: lane l -> col n0+(l&15); halves 0..7 = K[k0+ksel*8 .. +7], halves 8..15 = K[k0+16+ksel*8 .. +7]
__device__ __forceinline__ v16h load_tileB(const half_t* base, int stride,
                                           int n0, int k0, int lane) {
    const int  r    = n0 + (lane & 15);
    const int  ksel = lane >> 4;
    const half_t* p = base + r * stride + k0 + ksel * 8;
    v8h lo = *(const v8h*)(p);
    v8h hi = *(const v8h*)(p + 16);
    v16h b;
#pragma unroll
    for (int i = 0; i < 8; ++i) { b[i] = lo[i]; b[i + 8] = hi[i]; }
    return b;
}

// D f32 16x16: lane l covers N = n0+(l&15); VGPR r covers M = m0 + r + 8*(l>>4).
// Fuse bias + sin(w0*x), convert to f16, scatter to row-major LDS activations.
__device__ __forceinline__ void store_sin(half_t* base, int stride, int m0, int n0,
                                          int lane, const v8f& acc,
                                          const float* __restrict__ bias, float w0) {
    const int n  = n0 + (lane & 15);
    const float bn = bias[n];
    const int mb = m0 + ((lane >> 4) << 3);
#pragma unroll
    for (int r = 0; r < 8; ++r) {
        float v = __sinf(w0 * (acc[r] + bn));
        base[(mb + r) * stride + n] = (half_t)v;
    }
}

// ---------------------------------------------------------------------------
// Layer bodies (per workgroup of 8 waves x 16 rows = 128 rows)
// ---------------------------------------------------------------------------
#define SA_STRIDE 264   // 256 + 8 halves padding
#define A64_STRIDE 72   // 64 + 8 halves padding

// 256 -> 256 siren layer (w0 = 1)
__device__ void dense256(half_t* s_act, half_t* s_w, const half_t* __restrict__ wsrc,
                         const float* __restrict__ bias, int tid, int wave, int lane) {
    __syncthreads();   // previous users of s_w done
    for (int i = tid; i < 256 * 32; i += 256) {             // 256 rows x 32 chunks
        int r = i >> 5, c = (i & 31) << 3;
        *(v8h*)(s_w + r * SA_STRIDE + c) = *(const v8h*)(wsrc + (r << 8) + c);
    }
    __syncthreads();
    const int m0 = wave << 4;
    v16h A[8];
#pragma unroll
    for (int t = 0; t < 8; ++t) A[t] = load_tileA(s_act, SA_STRIDE, m0, t << 5, lane);
    for (int nt = 0; nt < 16; ++nt) {
        v8f acc = {};
#pragma unroll
        for (int k = 0; k < 8; ++k) {
            v16h B = load_tileB(s_w, SA_STRIDE, nt << 4, k << 5, lane);
            acc = WMMA_F16(A[k], B, acc);
        }
        store_sin(s_act, SA_STRIDE, m0, nt << 4, lane, acc, bias, 1.0f);
    }
}

// 256 -> 32 siren layer, result into columns 0..31 of the 64-wide buffer
__device__ void dense256to32(const half_t* s_act, half_t* s_a64, half_t* s_w,
                             const half_t* __restrict__ wsrc,
                             const float* __restrict__ bias, int tid, int wave, int lane) {
    __syncthreads();
    for (int i = tid; i < 32 * 32; i += 256) {
        int r = i >> 5, c = (i & 31) << 3;
        *(v8h*)(s_w + r * SA_STRIDE + c) = *(const v8h*)(wsrc + (r << 8) + c);
    }
    __syncthreads();
    const int m0 = wave << 4;
    v16h A[8];
#pragma unroll
    for (int t = 0; t < 8; ++t) A[t] = load_tileA(s_act, SA_STRIDE, m0, t << 5, lane);
#pragma unroll
    for (int nt = 0; nt < 2; ++nt) {
        v8f acc = {};
#pragma unroll
        for (int k = 0; k < 8; ++k) {
            v16h B = load_tileB(s_w, SA_STRIDE, nt << 4, k << 5, lane);
            acc = WMMA_F16(A[k], B, acc);
        }
        store_sin(s_a64, A64_STRIDE, m0, nt << 4, lane, acc, bias, 1.0f);
    }
}

// 64 -> 64 siren layer (rf stack)
__device__ void rf_dense64(half_t* s_a64, half_t* s_w, const half_t* __restrict__ wsrc,
                           const float* __restrict__ bias, int tid, int wave, int lane) {
    __syncthreads();
    for (int i = tid; i < 64 * 8; i += 256) {               // 64 rows x 8 chunks
        int r = i >> 3, c = (i & 7) << 3;
        *(v8h*)(s_w + r * A64_STRIDE + c) = *(const v8h*)(wsrc + (r << 6) + c);
    }
    __syncthreads();
    const int m0 = wave << 4;
    v16h A0 = load_tileA(s_a64, A64_STRIDE, m0, 0, lane);
    v16h A1 = load_tileA(s_a64, A64_STRIDE, m0, 32, lane);
#pragma unroll
    for (int nt = 0; nt < 4; ++nt) {
        v8f acc = {};
        v16h B0 = load_tileB(s_w, A64_STRIDE, nt << 4, 0, lane);
        acc = WMMA_F16(A0, B0, acc);
        v16h B1 = load_tileB(s_w, A64_STRIDE, nt << 4, 32, lane);
        acc = WMMA_F16(A1, B1, acc);
        store_sin(s_a64, A64_STRIDE, m0, nt << 4, lane, acc, bias, 1.0f);
    }
}

// ---------------------------------------------------------------------------
struct Params {
    const float* pn; const float* v;
    const float* w0; const float* b0;              // lm layer0 (6->256), f32
    const half_t* wh;                              // converted f16 weights in d_ws
    const float* b1; const float* b2; const float* b3; const float* b4; const float* b5;
    const float* b6;                               // lm_bs[6] (32)
    const float* vw; const float* vb;              // vup 3->32, f32
    const float* rb0; const float* rb1; const float* rb2; const float* rb3;
    const float* rw4; const float* rb4;            // final 64->3, f32
    float* out;
};

// d_ws layout (halves): lm_ws[1..5] @ i*65536, lm_ws[6] @ 327680, rf_ws[0..3] @ 335872+j*4096
#define OFF_LM(i)  ((size_t)(i) * 65536u)          // i = 0..4 for lm layers 1..5
#define OFF_LM6    327680u
#define OFF_RF(j)  (335872u + (size_t)(j) * 4096u)

__global__ __launch_bounds__(256) void siren_fused(Params p) {
    extern __shared__ __align__(16) char smem[];
    half_t* s_w   = (half_t*)smem;                                 // 256*264*2 = 135168 B
    half_t* s_act = (half_t*)(smem + 135168);                      // 128*264*2 =  67584 B
    half_t* s_a64 = (half_t*)(smem + 135168 + 67584);              // 128*72*2  =  18432 B

    const int tid  = threadIdx.x;
    const int wave = tid >> 5, lane = tid & 31;
    const int row0 = blockIdx.x * 128;

    // --- stage small f32 weights (layer0 + vup) into s_w's float view ---
    float* sf = (float*)s_w;   // [0,1536)=W0, [1536,1792)=b0, [1792,1888)=vupW, [1888,1920)=vupB
    for (int i = tid; i < 1920; i += 256) {
        float val;
        if (i < 1536)       val = p.w0[i];
        else if (i < 1792)  val = p.b0[i - 1536];
        else if (i < 1888)  val = p.vw[i - 1792];
        else                val = p.vb[i - 1888];
        sf[i] = val;
    }
    __syncthreads();

    // --- layer0: 6->256 siren w0=20, and vup: 3->32 siren w0=60 (scalar VALU) ---
    {
        const int m = tid >> 1;
        const float* pnr = p.pn + (size_t)(row0 + m) * 6;
        float x0 = pnr[0], x1 = pnr[1], x2 = pnr[2], x3 = pnr[3], x4 = pnr[4], x5 = pnr[5];
        const int nbase = (tid & 1) << 7;
        for (int j = 0; j < 128; ++j) {
            int n = nbase + j;
            const float* wr = sf + n * 6;
            float a = sf[1536 + n] + x0*wr[0] + x1*wr[1] + x2*wr[2]
                                   + x3*wr[3] + x4*wr[4] + x5*wr[5];
            s_act[m * SA_STRIDE + n] = (half_t)__sinf(20.0f * a);
        }
        const float* vr = p.v + (size_t)(row0 + m) * 3;
        float y0 = vr[0], y1 = vr[1], y2 = vr[2];
        const int cb = (tid & 1) << 4;
        for (int j = 0; j < 16; ++j) {
            int n = cb + j;
            const float* wr = sf + 1792 + n * 3;
            float a = sf[1888 + n] + y0*wr[0] + y1*wr[1] + y2*wr[2];
            s_a64[m * A64_STRIDE + 32 + n] = (half_t)__sinf(60.0f * a);
        }
    }

    // --- 5 hidden 256->256 WMMA layers ---
    dense256(s_act, s_w, p.wh + OFF_LM(0), p.b1, tid, wave, lane);
    dense256(s_act, s_w, p.wh + OFF_LM(1), p.b2, tid, wave, lane);
    dense256(s_act, s_w, p.wh + OFF_LM(2), p.b3, tid, wave, lane);
    dense256(s_act, s_w, p.wh + OFF_LM(3), p.b4, tid, wave, lane);
    dense256(s_act, s_w, p.wh + OFF_LM(4), p.b5, tid, wave, lane);

    // --- 256->32 into columns 0..31 of the 64-wide buffer ---
    dense256to32(s_act, s_a64, s_w, p.wh + OFF_LM6, p.b6, tid, wave, lane);

    // --- 4 rf 64->64 WMMA layers ---
    rf_dense64(s_a64, s_w, p.wh + OFF_RF(0), p.rb0, tid, wave, lane);
    rf_dense64(s_a64, s_w, p.wh + OFF_RF(1), p.rb1, tid, wave, lane);
    rf_dense64(s_a64, s_w, p.wh + OFF_RF(2), p.rb2, tid, wave, lane);
    rf_dense64(s_a64, s_w, p.wh + OFF_RF(3), p.rb3, tid, wave, lane);

    // --- final 64->3 + sigmoid (scalar), cross-wave reads -> barrier ---
    __syncthreads();
    if (tid < 128) {
        const int m = tid;
        const half_t* xr = s_a64 + m * A64_STRIDE;
#pragma unroll
        for (int c = 0; c < 3; ++c) {
            const float* wr = p.rw4 + c * 64;
            float acc = p.rb4[c];
            for (int k = 0; k < 64; ++k) acc += (float)xr[k] * wr[k];
            p.out[(size_t)(row0 + m) * 3 + c] = 1.0f / (1.0f + __expf(-acc));
        }
    }
}

__global__ void cvt_f32_f16(const float* __restrict__ src, half_t* __restrict__ dst, int n) {
    int i = blockIdx.x * blockDim.x + threadIdx.x;
    if (i < n) dst[i] = (half_t)src[i];
}

// ---------------------------------------------------------------------------
extern "C" void kernel_launch(void* const* d_in, const int* in_sizes, int n_in,
                              void* d_out, int out_size, void* d_ws, size_t ws_size,
                              hipStream_t stream) {
    (void)in_sizes; (void)n_in; (void)out_size; (void)ws_size;
    const int N = 131072;
    half_t* wh = (half_t*)d_ws;

    // Input order: 0 pn, 1 v, 2..8 lm_ws[0..6], 9..15 lm_bs[0..6],
    //              16 vup_w, 17 vup_b, 18..22 rf_ws[0..4], 23..27 rf_bs[0..4]
    auto cvt = [&](int idx, size_t off, int n) {
        cvt_f32_f16<<<(n + 255) / 256, 256, 0, stream>>>((const float*)d_in[idx], wh + off, n);
    };
    cvt(3, OFF_LM(0), 65536);  // lm_ws[1..5]
    cvt(4, OFF_LM(1), 65536);
    cvt(5, OFF_LM(2), 65536);
    cvt(6, OFF_LM(3), 65536);
    cvt(7, OFF_LM(4), 65536);
    cvt(8, OFF_LM6, 8192);     // lm_ws[6] 32x256
    cvt(18, OFF_RF(0), 4096);  // rf_ws[0..3] 64x64
    cvt(19, OFF_RF(1), 4096);
    cvt(20, OFF_RF(2), 4096);
    cvt(21, OFF_RF(3), 4096);

    Params p;
    p.pn = (const float*)d_in[0];  p.v  = (const float*)d_in[1];
    p.w0 = (const float*)d_in[2];  p.b0 = (const float*)d_in[9];
    p.wh = wh;
    p.b1 = (const float*)d_in[10]; p.b2 = (const float*)d_in[11];
    p.b3 = (const float*)d_in[12]; p.b4 = (const float*)d_in[13];
    p.b5 = (const float*)d_in[14]; p.b6 = (const float*)d_in[15];
    p.vw = (const float*)d_in[16]; p.vb = (const float*)d_in[17];
    p.rb0 = (const float*)d_in[23]; p.rb1 = (const float*)d_in[24];
    p.rb2 = (const float*)d_in[25]; p.rb3 = (const float*)d_in[26];
    p.rw4 = (const float*)d_in[22]; p.rb4 = (const float*)d_in[27];
    p.out = (float*)d_out;

    const size_t shmem = 135168 + 67584 + 18432;   // 221,184 B <= 320 KB/WG
    siren_fused<<<N / 128, 256, shmem, stream>>>(p);
}
